// Pseudo3DDCN_33749853012459
// MI455X (gfx1250) — matrix operand order repaired and checked by reading, not compiled, not run on hardware
//
#include <hip/hip_runtime.h>
#include <hip/hip_bf16.h>

typedef __attribute__((ext_vector_type(16))) _Float16 v16h;
typedef __attribute__((ext_vector_type(8)))  _Float16 v8h;
typedef __attribute__((ext_vector_type(2)))  _Float16 h2;
typedef __attribute__((ext_vector_type(8)))  float    v8f;

#define BB   2
#define CC   32
#define TT   8
#define HH   192
#define WW   192
#define HWp  (HH*WW)           // 36864
#define THW  (TT*HWp)          // 294912
#define KTAP 9
#define CHUNKS 96
#define TILES_PER_WAVE 3       // 96 chunks * 8 waves * 3 tiles * 16 px = 36864 px/slice

__device__ __forceinline__ int clampi(int v, int lo, int hi) {
    return v < lo ? lo : (v > hi ? hi : v);
}

// Pass 1: feat [B,C,T,H,W] f32 -> feat_t [slice=(b*8+t), H, W, C] f16.
// Each thread converts one channel pair and writes a packed half2 ->
// fully coalesced 128B stores; gathers on the read side hit L2.
__global__ void dcn_transpose_feat(const float* __restrict__ feat,
                                   _Float16* __restrict__ ft) {
    int idx   = blockIdx.x * blockDim.x + threadIdx.x;   // total B*T*HW*16
    int cp    = idx & 15;                                // channel pair
    int rest  = idx >> 4;
    int pix   = rest % HWp;
    int slice = rest / HWp;                              // b*8 + t
    int b = slice >> 3, t = slice & 7;
    const float* src = feat + (((size_t)b * CC + 2 * cp) * TT + t) * HWp + pix;
    h2 v;
    v[0] = (_Float16)src[0];
    v[1] = (_Float16)src[(size_t)THW];                   // next channel (+T*H*W)
    ((h2*)ft)[(size_t)slice * HWp * 16 + (size_t)pix * 16 + cp] = v;
}

// Pass 2: pre-swizzle weights into exact WMMA A-fragment layout.
// frag (t, mt, s): A[m=0..15][kk=0..31] = weight[t][o=mt*16+m][cin=kk][tap=s]
// A 16x32 layout used here (contiguous k, same as B): lane L holds row m=L%16; half h holds
// kk = h + (L>=16 ? 16 : 0)
__global__ void dcn_prep_afrag(const float* __restrict__ w,
                               _Float16* __restrict__ af) {
    int id   = blockIdx.x * blockDim.x + threadIdx.x;    // total 8*2*9*32*16 = 73728
    int h    = id & 15;
    int lane = (id >> 4) & 31;
    int s    = (id >> 9) % 9;
    int mt   = (id / (512 * 9)) & 1;
    int t    =  id / (512 * 18);
    int kk   = h + ((lane >> 4) * 16);
    int m    = lane & 15;
    int o    = mt * 16 + m;
    float val = w[(((size_t)t * CC + o) * CC + kk) * KTAP + s];
    af[((((size_t)t * 2 + mt) * KTAP + s) * 32 + lane) * 16 + h] = (_Float16)val;
}

// Main DCN kernel: one wave = 16 pixels x 32 outputs via 2 accumulators,
// 9 K-steps of v_wmma_f32_16x16x32_f16 (K = 32 input channels per tap).
// Bilinear sampling done in packed f16 straight into the B fragment.
__global__ void __launch_bounds__(256)
dcn_main(const _Float16* __restrict__ ftp, const float* __restrict__ off,
         const _Float16* __restrict__ af, float* __restrict__ out) {
    const int wave  = threadIdx.x >> 5;
    const int lane  = threadIdx.x & 31;
    const int chunk = blockIdx.x % CHUNKS;
    const int slice = blockIdx.x / CHUNKS;               // b*8 + t
    const int b = slice >> 3, t = slice & 7;

    const int n   = lane & 15;  // pixel within tile (B/D column)
    const int cg  = lane >> 4;  // 0: channels 0-15 / M rows 0-7; 1: 16-31 / 8-15
    const int cg2 = cg * 2;     // v8h chunk offset within a pixel's 32 channels

    const v8h*  fb   = (const v8h*)(ftp + (size_t)slice * HWp * CC);
    const v16h* afv  = (const v16h*)af;
    const int   afb0 = ((t * 2 + 0) * KTAP) * 32 + lane;
    const int   afb1 = ((t * 2 + 1) * KTAP) * 32 + lane;
    const size_t offB = ((size_t)b * 18 * TT + t) * HWp;

    const int tile0 = (chunk * 8 + wave) * TILES_PER_WAVE;
    for (int it = 0; it < TILES_PER_WAVE; ++it) {
        const int tile = tile0 + it;
        const int lin  = tile * 16;
        const int py   = lin / WW;       // WW % 16 == 0 -> whole tile in one row
        const int px   = lin % WW + n;
        const int pixOff = py * WW + px;

        // pre-folded sampling-grid constants: ys = dy + cy[ky], xs = dx + cx[kx]
        const float cy[3] = { (float)(py - 1), (float)py, (float)(py + 1) };
        const float cx[3] = { (float)(px - 1), (float)px, (float)(px + 1) };

        v8f acc0 = {};
        v8f acc1 = {};

        #pragma unroll
        for (int s = 0; s < KTAP; ++s) {
            // mmcv offset layout: channel 2s = dy, 2s+1 = dx (stride T*H*W)
            size_t oi = offB + (size_t)(2 * s) * THW + pixOff;
            float dy = off[oi];
            float dx = off[oi + (size_t)THW];

            float ys = dy + cy[s / 3];
            float xs = dx + cx[s % 3];
            float y0f = floorf(ys), x0f = floorf(xs);
            float fy = ys - y0f,    fx = xs - x0f;
            int y0 = (int)y0f, x0 = (int)x0f;
            int y1 = y0 + 1,   x1 = x0 + 1;

            // fold zero-padding validity into the 1D weights (4 cndmask)
            float wy0 = ((unsigned)y0 < (unsigned)HH) ? (1.0f - fy) : 0.0f;
            float wy1 = ((unsigned)y1 < (unsigned)HH) ? fy          : 0.0f;
            float wx0 = ((unsigned)x0 < (unsigned)WW) ? (1.0f - fx) : 0.0f;
            float wx1 = ((unsigned)x1 < (unsigned)WW) ? fx          : 0.0f;

            _Float16 h00 = (_Float16)(wy0 * wx0);
            _Float16 h01 = (_Float16)(wy0 * wx1);
            _Float16 h10 = (_Float16)(wy1 * wx0);
            _Float16 h11 = (_Float16)(wy1 * wx1);

            int yc0 = clampi(y0, 0, HH - 1), yc1 = clampi(y1, 0, HH - 1);
            int xc0 = clampi(x0, 0, WW - 1), xc1 = clampi(x1, 0, WW - 1);
            int r0 = yc0 * WW, r1 = yc1 * WW;
            // v8h-unit offsets within the slice (32-bit, enables saddr+voffset)
            int i00 = (r0 + xc0) * 4 + cg2;
            int i01 = (r0 + xc1) * 4 + cg2;
            int i10 = (r1 + xc0) * 4 + cg2;
            int i11 = (r1 + xc1) * 4 + cg2;

            v8h q0 = fb[i00],     q1 = fb[i01],     q2 = fb[i10],     q3 = fb[i11];
            v8h lo = q0 * h00 + q1 * h01 + q2 * h10 + q3 * h11;
            q0 = fb[i00 + 1]; q1 = fb[i01 + 1]; q2 = fb[i10 + 1]; q3 = fb[i11 + 1];
            v8h hi = q0 * h00 + q1 * h01 + q2 * h10 + q3 * h11;

            // B fragment: lane L holds column n=L%16, K rows cg*16 .. cg*16+15
            v16h bf = __builtin_shufflevector(lo, hi,
                0, 1, 2, 3, 4, 5, 6, 7, 8, 9, 10, 11, 12, 13, 14, 15);

            v16h a0 = afv[afb0 + s * 32];
            v16h a1 = afv[afb1 + s * 32];
            acc0 = __builtin_amdgcn_wmma_f32_16x16x32_f16(false, a0, false, bf,
                                                          (short)0, acc0, false, false);
            acc1 = __builtin_amdgcn_wmma_f32_16x16x32_f16(false, a1, false, bf,
                                                          (short)0, acc1, false, false);
        }

        // D layout: VGPR r -> M = r + 8*cg, column = n. out[b][o][t][y][x].
        size_t ob = (((size_t)b * CC) * TT + t) * HWp + pixOff;
        #pragma unroll
        for (int r = 0; r < 8; ++r) {
            int o0 = cg * 8 + r;
            out[ob + (size_t)o0 * THW]        = acc0[r];
            out[ob + (size_t)(o0 + 16) * THW] = acc1[r];
        }
    }
}

extern "C" void kernel_launch(void* const* d_in, const int* in_sizes, int n_in,
                              void* d_out, int out_size, void* d_ws, size_t ws_size,
                              hipStream_t stream) {
    const float* feat = (const float*)d_in[0];   // [2,32,8,192,192]
    const float* off  = (const float*)d_in[1];   // [2,18,8,192,192]
    const float* wgt  = (const float*)d_in[2];   // [8,32,32,3,3]
    float* out = (float*)d_out;                  // [2,32,8,192,192]

    // workspace: A-fragments (147456 B, 32B aligned) then f16 transposed feat
    _Float16* af = (_Float16*)d_ws;
    _Float16* ft = (_Float16*)((char*)d_ws + 147456);

    dcn_prep_afrag<<<73728 / 256, 256, 0, stream>>>(wgt, af);
    dcn_transpose_feat<<<9437184 / 256, 256, 0, stream>>>(feat, ft);
    dcn_main<<<(BB * TT) * CHUNKS, 256, 0, stream>>>(ft, off, af, out);
}
